// BahdanauAttention_16827681866015
// MI455X (gfx1250) — matrix operand order repaired and checked by reading, not compiled, not run on hardware
//
#include <hip/hip_runtime.h>
#include <math.h>

// ---- problem constants (match reference) ----
#define B_  32
#define S_  2048
#define D_  1024
#define U_  1024

// ---- score kernel tiling ----
#define BM       64      // rows of (b,s) staged per block
#define ASTRIDE  1040    // bf16 elements per staged LDS row (1024 + 16 pad -> breaks bank conflicts)

typedef __attribute__((ext_vector_type(16))) __bf16 v16bf;
typedef __attribute__((ext_vector_type(8)))  __bf16 v8bf;
typedef __attribute__((ext_vector_type(4)))  __bf16 v4bf;
typedef __attribute__((ext_vector_type(8)))  float  v8f;

// Native CDNA5 V_TANH_F32 (TRANS32 op, co-executes with XDL/WMMA) when the
// toolchain exposes it; fall back to libm expansion otherwise.
__device__ __forceinline__ float fast_tanh(float x) {
#if __has_builtin(__builtin_amdgcn_tanhf)
    return __builtin_amdgcn_tanhf(x);
#elif __has_builtin(__builtin_amdgcn_tanh_f32)
    return __builtin_amdgcn_tanh_f32(x);
#else
    return tanhf(x);
#endif
}

// ---------------------------------------------------------------------------
// Kernel 1: qb[b,u] = query[b,:] @ w1[:,u] + b1[u] + b2[u]
// (q_proj with both biases pre-folded; reused for every s)
// ---------------------------------------------------------------------------
__global__ __launch_bounds__(256)
void k_qproj(const float* __restrict__ q, const float* __restrict__ w1,
             const float* __restrict__ b1, const float* __restrict__ b2,
             float* __restrict__ qb) {
    int gid = blockIdx.x * 256 + threadIdx.x;      // [0, B_*U_)
    int b = gid >> 10;
    int u = gid & (U_ - 1);
    float acc = b1[u] + b2[u];
    const float* qr = q + b * D_;
#pragma unroll 8
    for (int d = 0; d < D_; ++d)
        acc += qr[d] * w1[(size_t)d * U_ + u];      // coalesced across u
    qb[gid] = acc;
}

// ---------------------------------------------------------------------------
// Kernel 2: w2t[u][d] = (bf16) w2[d][u]   (transposed bf16 copy, 2 MB, L2-resident)
// ---------------------------------------------------------------------------
__global__ __launch_bounds__(256)
void k_w2t(const float* __restrict__ w2, __bf16* __restrict__ w2t) {
    __shared__ float tile[32][33];
    int u0 = blockIdx.x * 32, d0 = blockIdx.y * 32;
    int tx = threadIdx.x, ty = threadIdx.y;        // block (32,8)
#pragma unroll
    for (int i = 0; i < 32; i += 8)
        tile[ty + i][tx] = w2[(size_t)(d0 + ty + i) * U_ + u0 + tx];
    __syncthreads();
#pragma unroll
    for (int i = 0; i < 32; i += 8)
        w2t[(size_t)(u0 + ty + i) * D_ + d0 + tx] = (__bf16)tile[tx][ty + i];
}

// ---------------------------------------------------------------------------
// Kernel 3: fused  scores[b,s] = wv . tanh(qb[b,:] + values[b,s,:] @ w2) + bv
// Block: 256 threads (8 waves), batch b = blockIdx.y, rows [s0, s0+BM).
// Wave w: row-tile rt = w%4 (16 rows), N-half nh = w/4 (N tiles with n%2==nh).
// ---------------------------------------------------------------------------
__global__ __launch_bounds__(256)
void k_score(const float* __restrict__ values, const __bf16* __restrict__ w2t,
             const float* __restrict__ qb, const float* __restrict__ wv,
             const float* __restrict__ bv, float* __restrict__ scores) {
    extern __shared__ __align__(16) char smem[];
    __bf16* Alds = (__bf16*)smem;                              // BM x ASTRIDE bf16
    float*  sred = (float*)(smem + (size_t)BM * ASTRIDE * 2);  // 64 floats

    const int tid  = threadIdx.x;
    const int bidx = blockIdx.y;
    const int s0   = blockIdx.x * BM;

    // ---- stage values[b, s0:s0+BM, :] -> LDS as bf16 (f32 read once from HBM) ----
    const float4* vrow = (const float4*)(values + ((size_t)bidx * S_ + s0) * D_);
    for (int i = tid; i < BM * (D_ / 4); i += 256) {
        int r = i >> 8;               // D_/4 == 256 float4 per row
        int c = i & 255;
        float4 f = vrow[(size_t)r * (D_ / 4) + c];
        v4bf o;
        o[0] = (__bf16)f.x; o[1] = (__bf16)f.y;
        o[2] = (__bf16)f.z; o[3] = (__bf16)f.w;
        *(v4bf*)(Alds + (size_t)r * ASTRIDE + c * 4) = o;
    }
    __syncthreads();

    const int lane   = tid & 31;
    const int wid    = tid >> 5;       // 0..7
    const int rt     = wid & 3;        // row tile within block
    const int nh     = wid >> 2;       // 0/1: which half of the N tiles
    const int halfId = lane >> 4;      // lane half (K split: half h holds K 16h..16h+15)
    const int l16    = lane & 15;
    const int arow   = rt * 16 + l16;

    // Per-lane A base: lane half 0 holds K {0..15},
    // half 1 holds K {16..31} (two adjacent 16B chunks, matching B).
    const __bf16* aptr = Alds + (size_t)arow * ASTRIDE + halfId * 16;

    float sacc[8];
#pragma unroll
    for (int v = 0; v < 8; ++v) sacc[v] = 0.f;

    for (int nt = nh; nt < U_ / 16; nt += 2) {
        const int ncol = nt * 16 + l16;
        // B 32x16 bf16 layout: lanes 0-15 hold K=kb..kb+15 of column ncol,
        // lanes 16-31 hold K=kb+16..kb+31  -> 32 contiguous bytes from w2t.
        const __bf16* bptr = w2t + (size_t)ncol * D_ + halfId * 16;

        v8f acc = {};
#pragma unroll 4
        for (int kb = 0; kb < D_; kb += 32) {
            v8bf alo = *(const v8bf*)(aptr + kb);        // k: +off .. +off+7
            v8bf ahi = *(const v8bf*)(aptr + kb + 8);    // k: +off+8 .. +off+15
            v16bf a;
#pragma unroll
            for (int e = 0; e < 8; ++e) { a[e] = alo[e]; a[e + 8] = ahi[e]; }
            v16bf bb = *(const v16bf*)(bptr + kb);
            acc = __builtin_amdgcn_wmma_f32_16x16x32_bf16(
                      false, a, false, bb, (short)0, acc, false, false);
        }

        // Fused epilogue on C fragment: lane holds N=ncol, VGPR v -> M = v + 8*halfId
        float qv  = qb[(size_t)bidx * U_ + ncol];
        float wvv = wv[ncol];
#pragma unroll
        for (int v = 0; v < 8; ++v)
            sacc[v] += fast_tanh(acc[v] + qv) * wvv;
    }

    // Reduce across the 16 lanes of each half (sums the 16 N-columns per tile)
#pragma unroll
    for (int off = 8; off >= 1; off >>= 1)
#pragma unroll
        for (int v = 0; v < 8; ++v)
            sacc[v] += __shfl_xor(sacc[v], off, 32);

    // Combine the two N-half waves via LDS, add bv, write raw scores
    if (nh == 0 && l16 == 0) {
#pragma unroll
        for (int v = 0; v < 8; ++v)
            sred[rt * 16 + halfId * 8 + v] = sacc[v];
    }
    __syncthreads();
    if (nh == 1 && l16 == 0) {
        float bvv = bv[0];
#pragma unroll
        for (int v = 0; v < 8; ++v) {
            int m = rt * 16 + halfId * 8 + v;
            scores[(size_t)bidx * S_ + s0 + m] = sacc[v] + sred[m] + bvv;
        }
    }
}

// ---------------------------------------------------------------------------
// Kernel 4: softmax over S per batch -> attention weights (output #2)
// ---------------------------------------------------------------------------
__global__ __launch_bounds__(256)
void k_softmax(const float* __restrict__ scores, float* __restrict__ attn) {
    __shared__ float red[256];
    int b = blockIdx.x, tid = threadIdx.x;
    const float* sr = scores + (size_t)b * S_;
    float* ar = attn + (size_t)b * S_;

    float mx = -3.4e38f;
    for (int s = tid; s < S_; s += 256) mx = fmaxf(mx, sr[s]);
    red[tid] = mx; __syncthreads();
    for (int o = 128; o > 0; o >>= 1) {
        if (tid < o) red[tid] = fmaxf(red[tid], red[tid + o]);
        __syncthreads();
    }
    mx = red[0]; __syncthreads();

    float sum = 0.f;
    for (int s = tid; s < S_; s += 256) {
        float e = __expf(sr[s] - mx);
        ar[s] = e;
        sum += e;
    }
    red[tid] = sum; __syncthreads();
    for (int o = 128; o > 0; o >>= 1) {
        if (tid < o) red[tid] += red[tid + o];
        __syncthreads();
    }
    float inv = 1.f / red[0];
    for (int s = tid; s < S_; s += 256) ar[s] *= inv;
}

// ---------------------------------------------------------------------------
// Kernel 5: context[b,d] = sum_s attn[b,s] * values[b,s,d]   (HBM-bound stream)
// ---------------------------------------------------------------------------
__global__ __launch_bounds__(256)
void k_context(const float* __restrict__ values, const float* __restrict__ attn,
               float* __restrict__ ctx) {
    __shared__ float aw[S_];
    int b = blockIdx.y, dc = blockIdx.x, tid = threadIdx.x;
    for (int s = tid; s < S_; s += 256) aw[s] = attn[(size_t)b * S_ + s];
    __syncthreads();

    int d = dc * 256 + tid;
    const float* vp = values + (size_t)b * S_ * D_ + d;
    float acc = 0.f;
#pragma unroll 8
    for (int s = 0; s < S_; ++s)
        acc += aw[s] * vp[(size_t)s * D_];
    ctx[(size_t)b * D_ + d] = acc;
}

// ---------------------------------------------------------------------------
extern "C" void kernel_launch(void* const* d_in, const int* in_sizes, int n_in,
                              void* d_out, int out_size, void* d_ws, size_t ws_size,
                              hipStream_t stream) {
    const float* query  = (const float*)d_in[0];  // [B, D]
    const float* values = (const float*)d_in[1];  // [B, S, D]
    const float* w1     = (const float*)d_in[2];  // [D, U]
    const float* b1     = (const float*)d_in[3];  // [U]
    const float* w2     = (const float*)d_in[4];  // [D, U]
    const float* b2     = (const float*)d_in[5];  // [U]
    const float* wv     = (const float*)d_in[6];  // [U, 1]
    const float* bv     = (const float*)d_in[7];  // [1]

    float* out  = (float*)d_out;
    float* ctx  = out;               // output 0: context_vector [B, D]
    float* attn = out + B_ * D_;     // output 1: attention_weights [B, S, 1]

    char* ws = (char*)d_ws;
    float*  qb     = (float*)ws;                                  // 128 KB
    __bf16* w2t    = (__bf16*)(ws + 131072);                      // 2 MB
    float*  scores = (float*)(ws + 131072 + 2097152);             // 256 KB

    k_qproj<<<(B_ * U_) / 256, 256, 0, stream>>>(query, w1, b1, b2, qb);
    k_w2t<<<dim3(U_ / 32, D_ / 32), dim3(32, 8), 0, stream>>>(w2, w2t);

    size_t smem = (size_t)BM * ASTRIDE * sizeof(__bf16) + 64 * sizeof(float);
    k_score<<<dim3(S_ / BM, B_), 256, smem, stream>>>(values, w2t, qb, wv, bv, scores);

    k_softmax<<<B_, 256, 0, stream>>>(scores, attn);
    k_context<<<dim3(D_ / 256, B_), 256, 0, stream>>>(values, attn, ctx);
}
